// GraphScaledDotProductAttentionWithMask_6262062318238
// MI455X (gfx1250) — matrix operand order repaired and checked by reading, not compiled, not run on hardware
//
#include <hip/hip_runtime.h>
#include <hip/hip_bf16.h>
#include <math.h>

// ---- problem constants (match reference) ----
#define BSZ 2
#define NH 16
#define LQ 1024
#define LK 1024
#define DK 64
#define DV 64
#define DM (NH * DV)      // 1024
#define BH (BSZ * NH)     // 32
#define SP 1032           // padded LDS row stride for score buffer

typedef __attribute__((ext_vector_type(16))) __bf16 v16bf;
typedef __attribute__((ext_vector_type(8)))  float  v8f;

union BF16Frag { v16bf v; __bf16 e[16]; };
union AccFrag  { v8f  v;  float  e[8];  };

static __device__ __forceinline__ v8f wmma_bf16(const v16bf& a, const v16bf& b, const v8f& c) {
    return __builtin_amdgcn_wmma_f32_16x16x32_bf16(false, a, false, b, (short)0, c, false, false);
}

static __device__ __forceinline__ void split_bf16(float f, __bf16& hi, __bf16& lo) {
    hi = (__bf16)f;
    lo = (__bf16)(f - (float)hi);
}

// ---------------------------------------------------------------------------
// Kernel 1: predicted position per query (tiny MLP, VALU)
// pos = sigmoid( tanh(scaled_q @ Wv + bv) @ Ws + bs ) * (LK-1)
// ---------------------------------------------------------------------------
__global__ __launch_bounds__(256) void pos_kernel(
    const float* __restrict__ q, const float* __restrict__ Wv,
    const float* __restrict__ bv, const float* __restrict__ Ws,
    const float* __restrict__ bs, float* __restrict__ pos_out) {
    __shared__ float sWv[DK * DV];
    __shared__ float sWs[DV];
    __shared__ float sbv[DV];
    for (int i = threadIdx.x; i < DK * DV; i += blockDim.x) sWv[i] = Wv[i];
    for (int i = threadIdx.x; i < DV; i += blockDim.x) { sWs[i] = Ws[i]; sbv[i] = bv[i]; }
    __syncthreads();

    int qi = blockIdx.x * blockDim.x + threadIdx.x;   // [0, BH*LQ)
    if (qi >= BH * LQ) return;
    const float* qr = q + (size_t)qi * DK;
    float sq[DK];
#pragma unroll
    for (int d = 0; d < DK; ++d) sq[d] = qr[d] * 0.125f;   // 1/sqrt(64)

    float s = bs[0];
    for (int j = 0; j < DV; ++j) {
        float a = sbv[j];
#pragma unroll 8
        for (int d = 0; d < DK; ++d) a += sq[d] * sWv[d * DV + j];
        s += tanhf(a) * sWs[j];
    }
    float sig = 1.f / (1.f + __expf(-s));
    pos_out[qi] = sig * (float)(LK - 1);
}

// ---------------------------------------------------------------------------
// Kernel 2: fused attention for one (b*h, 16-query) tile.
//   scores = split-bf16 WMMA QK^T + bias + gaussian(pos-gather)   -> LDS
//   softmax (wave-shuffle reductions), weights -> global, P(bf16) -> LDS
//   O = P @ V via WMMA (V split hi/lo), written as [b, q, h, dv]
// ---------------------------------------------------------------------------
__global__ __launch_bounds__(128) void attn_kernel(
    const float* __restrict__ q, const float* __restrict__ k,
    const float* __restrict__ v, const float* __restrict__ bias,
    const float* __restrict__ gab, const float* __restrict__ pos,
    float* __restrict__ weights_out, float* __restrict__ x_out) {

    extern __shared__ char smem[];
    float*  S      = (float*)smem;                              // [16][SP] f32 scores
    __bf16* P      = (__bf16*)(smem + 16 * SP * sizeof(float)); // [16][LK] bf16 probs
    float*  posrow = (float*)(smem + 16 * SP * sizeof(float) + 16 * LK * sizeof(__bf16));

    const int bh    = blockIdx.x;      // b*NH + h
    const int qt    = blockIdx.y;      // 16-query tile index
    const int tid   = threadIdx.x;
    const int wave  = tid >> 5;        // 4 waves
    const int lane  = tid & 31;
    const int m16   = lane & 15;
    const int hhalf = lane >> 4;

    if (tid < 16) posrow[tid] = pos[(size_t)bh * LQ + qt * 16 + tid];
    __syncthreads();

    // ---- Q A-fragments (16x32 bf16 layout), hi/lo split, 2 K-chunks of 32 ----
    BF16Frag Ahi[2], Alo[2];
    const float* qrow = q + ((size_t)bh * LQ + qt * 16 + m16) * DK;
#pragma unroll
    for (int c = 0; c < 2; ++c) {
        int base = c * 32 + hhalf * 8;
#pragma unroll
        for (int e = 0; e < 16; ++e) {
            int kk = base + e + ((e >= 8) ? 8 : 0);
            split_bf16(qrow[kk] * 0.125f, Ahi[c].e[e], Alo[c].e[e]);
        }
    }

    // ---- QK^T over this wave's 256-key span, 16 keys per tile ----
    for (int t = wave * 16; t < wave * 16 + 16; ++t) {
        AccFrag acc;
#pragma unroll
        for (int i = 0; i < 8; ++i) acc.e[i] = 0.f;

        const float* krow0 = k + ((size_t)bh * LK + t * 16 + m16) * DK;
        __builtin_prefetch(krow0 + 16 * DK, 0, 1);   // next key tile -> global_prefetch
#pragma unroll
        for (int c = 0; c < 2; ++c) {
            BF16Frag Bhi, Blo;
            const float* kp = krow0 + c * 32 + hhalf * 8;
#pragma unroll
            for (int e = 0; e < 16; ++e) split_bf16(kp[e + ((e >= 8) ? 8 : 0)], Bhi.e[e], Blo.e[e]);
            acc.v = wmma_bf16(Ahi[c].v, Bhi.v, acc.v);
            acc.v = wmma_bf16(Alo[c].v, Bhi.v, acc.v);
            acc.v = wmma_bf16(Ahi[c].v, Blo.v, acc.v);
        }

        // epilogue: + bias + gaussian(pos) -> LDS scores
        const float* gr = gab + (size_t)bh * LK * LK;
#pragma unroll
        for (int r = 0; r < 8; ++r) {
            int m   = r + 8 * hhalf;           // row within tile
            int col = t * 16 + m16;            // key index
            float p  = posrow[m];
            float pu = ceilf(p), pd = floorf(p);
            float mu = gr[(size_t)(int)pu * LK + col];
            float md = gr[(size_t)(int)pd * LK + col];
            float sel = mu * (1.f - (pu - p)) + md * (1.f - (p - pd));
            float a = acc.e[r]
                    + bias[((size_t)bh * LQ + qt * 16 + m) * LK + col]
                    - 0.5f * sel * sel;        // (0.5*POS_WIN)^2 == 1
            S[m * SP + col] = a;
        }
    }
    __syncthreads();

    // ---- softmax: each wave owns 4 rows; shuffle reductions over 32 lanes ----
    for (int rr = 0; rr < 4; ++rr) {
        int m = wave * 4 + rr;
        float mx = -3.402823466e38f;
        for (int c2 = lane; c2 < LK; c2 += 32) mx = fmaxf(mx, S[m * SP + c2]);
#pragma unroll
        for (int o = 16; o >= 1; o >>= 1) mx = fmaxf(mx, __shfl_xor(mx, o, 32));
        float sum = 0.f;
        for (int c2 = lane; c2 < LK; c2 += 32) {
            float e2 = __expf(S[m * SP + c2] - mx);
            S[m * SP + c2] = e2;
            sum += e2;
        }
#pragma unroll
        for (int o = 16; o >= 1; o >>= 1) sum += __shfl_xor(sum, o, 32);
        float inv = 1.f / sum;
        float* wrow = weights_out + ((size_t)bh * LQ + qt * 16 + m) * LK;
        for (int c2 = lane; c2 < LK; c2 += 32) {
            float pw = S[m * SP + c2] * inv;
            wrow[c2] = pw;                 // softmax weights output
            P[m * LK + c2] = (__bf16)pw;   // bf16 probs for P@V
        }
    }
    __syncthreads();

    // ---- O = P @ V : wave owns dv columns [wave*16, wave*16+16) ----
    AccFrag o;
#pragma unroll
    for (int i = 0; i < 8; ++i) o.e[i] = 0.f;
    for (int kc = 0; kc < LK / 32; ++kc) {
        BF16Frag Ap;
        int abase = kc * 32 + hhalf * 8;
#pragma unroll
        for (int e = 0; e < 16; ++e) {
            int key = abase + e + ((e >= 8) ? 8 : 0);
            Ap.e[e] = P[m16 * LK + key];
        }
        BF16Frag Vhi, Vlo;
        int kb = kc * 32 + hhalf * 8;
        const float* vp = v + ((size_t)bh * LK + kb) * DV + wave * 16 + m16;
#pragma unroll
        for (int e = 0; e < 16; ++e) split_bf16(vp[(size_t)(e + ((e >= 8) ? 8 : 0)) * DV], Vhi.e[e], Vlo.e[e]);
        o.v = wmma_bf16(Ap.v, Vhi.v, o.v);
        o.v = wmma_bf16(Ap.v, Vlo.v, o.v);
    }
    // write as X[b, q, h, dv] so out-projection is a plain row-major GEMM
    int b = bh >> 4, h = bh & 15;
#pragma unroll
    for (int r = 0; r < 8; ++r) {
        int qrow_i = qt * 16 + r + 8 * hhalf;
        int dv     = wave * 16 + m16;
        x_out[(((size_t)b * LQ + qrow_i) * NH + h) * DV + dv] = o.e[r];
    }
}

// ---------------------------------------------------------------------------
// Kernel 3: out = X[2048,1024] @ W_out[1024,1024] + b_out  (split-bf16 WMMA)
// ---------------------------------------------------------------------------
__global__ __launch_bounds__(128) void outproj_kernel(
    const float* __restrict__ X, const float* __restrict__ W,
    const float* __restrict__ b, float* __restrict__ out) {
    const int mt    = blockIdx.x;                       // 16-row tile
    const int cb    = blockIdx.y * 64 + (threadIdx.x >> 5) * 16;  // 16-col tile per wave
    const int lane  = threadIdx.x & 31;
    const int m16   = lane & 15;
    const int hhalf = lane >> 4;

    AccFrag acc;
#pragma unroll
    for (int i = 0; i < 8; ++i) acc.e[i] = 0.f;

    const float* xrow = X + ((size_t)mt * 16 + m16) * DM;
    for (int kc = 0; kc < DM / 32; ++kc) {
        BF16Frag Ahi, Alo;
        int base = kc * 32 + hhalf * 8;
#pragma unroll
        for (int e = 0; e < 16; ++e) {
            int kk = base + e + ((e >= 8) ? 8 : 0);
            split_bf16(xrow[kk], Ahi.e[e], Alo.e[e]);
        }
        BF16Frag Bhi, Blo;
        int kb = kc * 32 + hhalf * 8;
        const float* wp = W + (size_t)kb * DM + cb + m16;
#pragma unroll
        for (int e = 0; e < 16; ++e) split_bf16(wp[(size_t)(e + ((e >= 8) ? 8 : 0)) * DM], Bhi.e[e], Blo.e[e]);
        acc.v = wmma_bf16(Ahi.v, Bhi.v, acc.v);
        acc.v = wmma_bf16(Alo.v, Bhi.v, acc.v);
        acc.v = wmma_bf16(Ahi.v, Blo.v, acc.v);
    }
#pragma unroll
    for (int r = 0; r < 8; ++r) {
        int row = mt * 16 + r + 8 * hhalf;
        int col = cb + m16;
        out[(size_t)row * DM + col] = acc.e[r] + b[col];
    }
}

// ---------------------------------------------------------------------------
extern "C" void kernel_launch(void* const* d_in, const int* in_sizes, int n_in,
                              void* d_out, int out_size, void* d_ws, size_t ws_size,
                              hipStream_t stream) {
    (void)in_sizes; (void)n_in; (void)out_size; (void)ws_size;
    const float* q    = (const float*)d_in[0];
    const float* k    = (const float*)d_in[1];
    const float* v    = (const float*)d_in[2];
    const float* bias = (const float*)d_in[3];
    const float* gab  = (const float*)d_in[4];
    const float* Wv   = (const float*)d_in[5];
    const float* bv   = (const float*)d_in[6];
    const float* Ws   = (const float*)d_in[7];
    const float* bs   = (const float*)d_in[8];
    const float* Wout = (const float*)d_in[9];
    const float* bout = (const float*)d_in[10];

    float* gout = (float*)d_out;                               // [2,1024,1024]
    float* wout = gout + (size_t)BSZ * LQ * DM;                // [2,16,1024,1024]

    float* pos_ws = (float*)d_ws;                              // BH*LQ floats
    float* x_ws   = (float*)((char*)d_ws + (size_t)BH * LQ * sizeof(float));  // [2048,1024]

    // 1) per-query predicted position
    pos_kernel<<<dim3((BH * LQ + 255) / 256), dim3(256), 0, stream>>>(
        q, Wv, bv, Ws, bs, pos_ws);

    // 2) fused attention (scores + bias + gaussian, softmax, P@V)
    size_t smem = 16 * SP * sizeof(float) + 16 * LK * sizeof(__bf16) + 16 * sizeof(float);
    attn_kernel<<<dim3(BH, LQ / 16), dim3(128), smem, stream>>>(
        q, k, v, bias, gab, pos_ws, wout, x_ws);

    // 3) output projection
    outproj_kernel<<<dim3((BSZ * LQ) / 16, DM / 64), dim3(128), 0, stream>>>(
        x_ws, Wout, bout, gout);
}
